// FastSpeech_37443524887208
// MI455X (gfx1250) — matrix operand order repaired and checked by reading, not compiled, not run on hardware
//
#include <hip/hip_runtime.h>
#include <hip/hip_bf16.h>

// ---------------------------------------------------------------------------
// FastSpeech forward on gfx1250 (MI455X).
// All GEMM-shaped work runs through a WMMA f16->f32 tiled kernel
// (v_wmma_f32_16x16x32_f16): 64x64x64 block tiles, 4 waves, each wave 32x32,
// 8 WMMAs per K-step. f32 master activations; f32->f16 conversion happens
// while staging tiles into LDS. B tile stored N-major in LDS so fragments are
// contiguous ds_load_b128s. Interior tiles take branch-free float4 paths.
// ---------------------------------------------------------------------------

typedef __attribute__((ext_vector_type(16))) _Float16 v16h;
typedef __attribute__((ext_vector_type(8)))  _Float16 v8h;
typedef __attribute__((ext_vector_type(8)))  float    v8f;

#define BM 64
#define BN 64
#define BK 64
#define SA 72   // LDS row stride (halves) for As[BM][BK]
#define SB 72   // LDS row stride (halves) for BsT[BN][BK]

#define GF_BT     1   // B given as [N,K] row-major (transposed)
#define GF_RELUO  2   // relu on output (after bias/resid)
#define GF_RELUA  4   // relu applied to A elements on load

// C[m,n] = epilogue( alpha * sum_k A[m+rowShift, k] * B(k,n) )
// A rows outside [0, rowsValid) read as zero (conv SAME padding).
// gridDim.z batches with per-z element strides aB/bB/cB/rB.
__launch_bounds__(128)
__global__ void gemm_wmma_f16(const float* __restrict__ A,
                              const float* __restrict__ Bm,
                              const float* __restrict__ bias,
                              const float* __restrict__ resid,
                              float* __restrict__ C,
                              int M, int N, int K, int lda, int ldb, int ldc,
                              int rowShift, int rowsValid,
                              long long aB, long long bB, long long cB, long long rB,
                              float alpha, int flags)
{
    __shared__ _Float16 As[BM * SA];
    __shared__ _Float16 BsT[BN * SB];

    const long long z = blockIdx.z;
    A  += z * aB;
    Bm += z * bB;
    C  += z * cB;
    if (resid) resid += z * rB;

    const int tile_m = blockIdx.y * BM;
    const int tile_n = blockIdx.x * BN;
    const int tid  = threadIdx.x;
    const int wave = tid >> 5;
    const int lane = tid & 31;
    const int wm = (wave >> 1) * 32;     // wave row offset in 64x64 tile
    const int wn = (wave & 1) * 32;      // wave col offset
    const bool bt    = (flags & GF_BT)    != 0;
    const bool reluA = (flags & GF_RELUA) != 0;

    // WMMA lane decomposition (CDNA5 ISA 7.12.2)
    const int lm = lane & 15;            // A row / C col within fragment
    const int kg = (lane >> 4) << 3;     // A K-group base: 0 or 8
    const int bk = (lane >> 4) << 4;     // unused; B fragments use kg (0 or 8), as A does

    // Tile-interior predicates (branch-free fast paths)
    const bool aRows = (tile_m + rowShift >= 0) &&
                       (tile_m + BM + rowShift <= rowsValid) &&
                       (tile_m + BM <= M);
    const bool bCols = (tile_n + BN <= N);

    v8f acc[2][2] = {};

    for (int k0 = 0; k0 < K; k0 += BK) {
        const bool kin = (k0 + BK <= K);

        // ---------------- stage A tile (64x64 f32 -> f16) ----------------
        if (aRows && kin) {
            #pragma unroll
            for (int i = 0; i < 8; ++i) {
                int f = tid + i * 128;             // float4 index 0..1023
                int r = f >> 4, c4 = (f & 15) << 2;
                const float4 v = *(const float4*)&A[(long long)(tile_m + r + rowShift) * lda + k0 + c4];
                _Float16* d = &As[r * SA + c4];
                float x0 = v.x, x1 = v.y, x2 = v.z, x3 = v.w;
                if (reluA) { x0 = fmaxf(x0, 0.f); x1 = fmaxf(x1, 0.f);
                             x2 = fmaxf(x2, 0.f); x3 = fmaxf(x3, 0.f); }
                d[0] = (_Float16)x0; d[1] = (_Float16)x1;
                d[2] = (_Float16)x2; d[3] = (_Float16)x3;
            }
        } else {
            #pragma unroll
            for (int i = 0; i < 8; ++i) {
                int f = tid + i * 128;
                int r = f >> 4, c4 = (f & 15) << 2;
                int gm = tile_m + r + rowShift;
                bool rowOk = ((unsigned)gm < (unsigned)rowsValid) && (tile_m + r) < M;
                _Float16* d = &As[r * SA + c4];
                #pragma unroll
                for (int e = 0; e < 4; ++e) {
                    int gk = k0 + c4 + e;
                    float x = (rowOk && gk < K) ? A[(long long)gm * lda + gk] : 0.f;
                    if (reluA) x = fmaxf(x, 0.f);
                    d[e] = (_Float16)x;
                }
            }
        }

        // ------------- stage B tile into N-major LDS (BsT[n][k]) ----------
        if (bt) {
            if (bCols && kin) {
                #pragma unroll
                for (int i = 0; i < 8; ++i) {
                    int f = tid + i * 128;
                    int r = f >> 4, c4 = (f & 15) << 2;   // r = n, c4 = k
                    const float4 v = *(const float4*)&Bm[(long long)(tile_n + r) * ldb + k0 + c4];
                    _Float16* d = &BsT[r * SB + c4];
                    d[0] = (_Float16)v.x; d[1] = (_Float16)v.y;
                    d[2] = (_Float16)v.z; d[3] = (_Float16)v.w;
                }
            } else {
                #pragma unroll
                for (int i = 0; i < 8; ++i) {
                    int f = tid + i * 128;
                    int r = f >> 4, c4 = (f & 15) << 2;
                    int gn = tile_n + r;
                    _Float16* d = &BsT[r * SB + c4];
                    #pragma unroll
                    for (int e = 0; e < 4; ++e) {
                        int gk = k0 + c4 + e;
                        d[e] = (_Float16)((gn < N && gk < K)
                                          ? Bm[(long long)gn * ldb + gk] : 0.f);
                    }
                }
            }
        } else {
            if (bCols && kin) {
                #pragma unroll
                for (int i = 0; i < 8; ++i) {
                    int f = tid + i * 128;
                    int r = f >> 4, c4 = (f & 15) << 2;   // r = k, c4 = n
                    const float4 v = *(const float4*)&Bm[(long long)(k0 + r) * ldb + tile_n + c4];
                    BsT[(c4 + 0) * SB + r] = (_Float16)v.x;
                    BsT[(c4 + 1) * SB + r] = (_Float16)v.y;
                    BsT[(c4 + 2) * SB + r] = (_Float16)v.z;
                    BsT[(c4 + 3) * SB + r] = (_Float16)v.w;
                }
            } else {
                #pragma unroll
                for (int i = 0; i < 8; ++i) {
                    int f = tid + i * 128;
                    int r = f >> 4, c4 = (f & 15) << 2;
                    int gk = k0 + r;
                    #pragma unroll
                    for (int e = 0; e < 4; ++e) {
                        int gn = tile_n + c4 + e;
                        BsT[(c4 + e) * SB + r] =
                            (_Float16)((gk < K && gn < N)
                                       ? Bm[(long long)gk * ldb + gn] : 0.f);
                    }
                }
            }
        }

        // prefetch next K tile while this one is consumed (global_prefetch_b8)
        if (k0 + BK < K) {
            int pr = tid >> 4, pc = (tid & 15) << 2;
            int gm = tile_m + pr + rowShift;
            if ((unsigned)gm < (unsigned)rowsValid)
                __builtin_prefetch(&A[(long long)gm * lda + k0 + BK + pc], 0, 0);
            if (bt) {
                if (tile_n + pr < N)
                    __builtin_prefetch(&Bm[(long long)(tile_n + pr) * ldb + k0 + BK + pc], 0, 0);
            } else {
                if (k0 + BK + pr < K)
                    __builtin_prefetch(&Bm[(long long)(k0 + BK + pr) * ldb + tile_n + pc], 0, 0);
            }
        }
        __syncthreads();

        // -------- fragments: all contiguous 16B LDS loads (ds_load_b128) --
        v16h af[2][2], bf[2][2];
        #pragma unroll
        for (int i = 0; i < 2; ++i) {
            #pragma unroll
            for (int kc = 0; kc < 2; ++kc) {
                const _Float16* p = &As[(wm + i * 16 + lm) * SA + kc * 32 + kg];
                v8h lo = *(const v8h*)p;
                v8h hi = *(const v8h*)(p + 16);
                #pragma unroll
                for (int e = 0; e < 8; ++e) { af[i][kc][e] = lo[e]; af[i][kc][8 + e] = hi[e]; }
            }
        }
        #pragma unroll
        for (int j = 0; j < 2; ++j) {
            #pragma unroll
            for (int kc = 0; kc < 2; ++kc) {
                const _Float16* p = &BsT[(wn + j * 16 + lm) * SB + kc * 32 + kg];
                v8h lo = *(const v8h*)p;
                v8h hi = *(const v8h*)(p + 16);
                #pragma unroll
                for (int e = 0; e < 8; ++e) { bf[j][kc][e] = lo[e]; bf[j][kc][8 + e] = hi[e]; }
            }
        }
        #pragma unroll
        for (int kc = 0; kc < 2; ++kc)
            #pragma unroll
            for (int i = 0; i < 2; ++i)
                #pragma unroll
                for (int j = 0; j < 2; ++j)
                    acc[i][j] = __builtin_amdgcn_wmma_f32_16x16x32_f16(
                        false, af[i][kc], false, bf[j][kc],
                        (short)0, acc[i][j], false, false);
        __syncthreads();
    }

    // ------------------------------ epilogue ------------------------------
    const bool reluO = (flags & GF_RELUO) != 0;
    const bool cFull = (tile_m + BM <= M) && (tile_n + BN <= N);
    const int mrow = ((lane >> 4) << 3);
    #pragma unroll
    for (int j = 0; j < 2; ++j) {
        const int n = tile_n + wn + j * 16 + lm;
        const float bb = (bias && n < N) ? bias[n] : 0.f;
        #pragma unroll
        for (int i = 0; i < 2; ++i) {
            const int mb = tile_m + wm + i * 16 + mrow;
            #pragma unroll
            for (int r = 0; r < 8; ++r) {
                const int m = mb + r;
                if (cFull || (m < M && n < N)) {
                    float v = acc[i][j][r] * alpha + bb;
                    if (resid) v += resid[(long long)m * ldc + n];
                    if (reluO && v < 0.f) v = 0.f;
                    C[(long long)m * ldc + n] = v;
                }
            }
        }
    }
}

// Row softmax with key-padding mask (tok[k]==0 -> score := -1e9), in place.
__global__ void softmax_mask(float* __restrict__ s, const int* __restrict__ tok, int Tk)
{
    float* row = s + (long long)blockIdx.x * Tk;
    __shared__ float red[256];
    float mx = -3.4e38f;
    for (int k = threadIdx.x; k < Tk; k += 256) {
        float v = (tok[k] == 0) ? -1e9f : row[k];
        mx = fmaxf(mx, v);
    }
    red[threadIdx.x] = mx; __syncthreads();
    for (int st = 128; st > 0; st >>= 1) {
        if (threadIdx.x < st) red[threadIdx.x] = fmaxf(red[threadIdx.x], red[threadIdx.x + st]);
        __syncthreads();
    }
    mx = red[0]; __syncthreads();
    float sum = 0.f;
    for (int k = threadIdx.x; k < Tk; k += 256) {
        float v = (tok[k] == 0) ? -1e9f : row[k];
        v = expf(v - mx);
        row[k] = v; sum += v;
    }
    red[threadIdx.x] = sum; __syncthreads();
    for (int st = 128; st > 0; st >>= 1) {
        if (threadIdx.x < st) red[threadIdx.x] += red[threadIdx.x + st];
        __syncthreads();
    }
    float inv = 1.0f / red[0];
    for (int k = threadIdx.x; k < Tk; k += 256) row[k] *= inv;
}

// LayerNorm over C columns; optional pad-mask (tok[row]==0 -> output 0).
__global__ void layernorm_mask(const float* __restrict__ in, float* __restrict__ out,
                               const float* __restrict__ g, const float* __restrict__ b,
                               const int* __restrict__ tok, int C)
{
    long long row = blockIdx.x;
    const float* x = in + row * C;
    float* y = out + row * C;
    __shared__ float red[256];
    float s = 0.f;
    for (int i = threadIdx.x; i < C; i += 256) s += x[i];
    red[threadIdx.x] = s; __syncthreads();
    for (int st = 128; st > 0; st >>= 1) {
        if (threadIdx.x < st) red[threadIdx.x] += red[threadIdx.x + st];
        __syncthreads();
    }
    float mean = red[0] / C; __syncthreads();
    s = 0.f;
    for (int i = threadIdx.x; i < C; i += 256) { float d = x[i] - mean; s += d * d; }
    red[threadIdx.x] = s; __syncthreads();
    for (int st = 128; st > 0; st >>= 1) {
        if (threadIdx.x < st) red[threadIdx.x] += red[threadIdx.x + st];
        __syncthreads();
    }
    float inv = rsqrtf(red[0] / C + 1e-5f);
    float msk = (tok && tok[row] == 0) ? 0.f : 1.f;
    for (int i = threadIdx.x; i < C; i += 256)
        y[i] = ((x[i] - mean) * inv * g[i] + b[i]) * msk;
}

// x[row] = src_emb[seq[row]] + pos_tab[pos[row]]
__global__ void embed_add(const int* __restrict__ seq, const int* __restrict__ pos,
                          const float* __restrict__ emb, const float* __restrict__ ptab,
                          float* __restrict__ out, int D)
{
    long long row = blockIdx.x;
    long long te = seq[row], tp = pos[row];
    for (int i = threadIdx.x; i < D; i += 256)
        out[row * D + i] = emb[te * D + i] + ptab[tp * D + i];
}

// y[row] += ptab[pos[row]]
__global__ void pos_add(const int* __restrict__ pos, const float* __restrict__ ptab,
                        float* __restrict__ y, int D)
{
    long long row = blockIdx.x;
    long long tp = pos[row];
    for (int i = threadIdx.x; i < D; i += 256)
        y[row * D + i] += ptab[tp * D + i];
}

// Length regulator: each (b,s) owns mel rows [start, start+dur); disjoint -> no races.
__global__ void length_reg(const int* __restrict__ durs, const float* __restrict__ x,
                           float* __restrict__ y, int TSRC, int MELMAX, int D)
{
    int b = blockIdx.y, s = blockIdx.x;
    const int* db = durs + (long long)b * TSRC;
    __shared__ int sh[2];
    if (threadIdx.x == 0) {
        int st = 0;
        for (int i = 0; i < s; ++i) st += db[i];
        sh[0] = st; sh[1] = db[s];
    }
    __syncthreads();
    int start = sh[0], d = sh[1];
    const float* src = x + ((long long)b * TSRC + s) * D;
    for (int r = 0; r < d; ++r) {
        int m = start + r;
        if (m >= MELMAX) break;
        float* dst = y + ((long long)b * MELMAX + m) * D;
        for (int i = threadIdx.x; i < D; i += 256) dst[i] = src[i];
    }
}

// out[row] = dot(h[row,:], w) + b0   (duration head, N=1)
__global__ void row_dot(const float* __restrict__ h, const float* __restrict__ w,
                        const float* __restrict__ b0, float* __restrict__ out, int C)
{
    long long row = blockIdx.x;
    __shared__ float red[256];
    float s = 0.f;
    for (int i = threadIdx.x; i < C; i += 256) s += h[row * C + i] * w[i];
    red[threadIdx.x] = s; __syncthreads();
    for (int st = 128; st > 0; st >>= 1) {
        if (threadIdx.x < st) red[threadIdx.x] += red[threadIdx.x + st];
        __syncthreads();
    }
    if (threadIdx.x == 0) out[row] = red[0] + b0[0];
}

// ---------------------------------------------------------------------------

struct FFTL {
    const float *wq, *wk, *wv, *wo, *bo, *ln1g, *ln1b;
    const float *c1w, *c1b, *c2w, *c2b, *ln2g, *ln2b;
};

extern "C" void kernel_launch(void* const* d_in, const int* in_sizes, int n_in,
                              void* d_out, int out_size, void* d_ws, size_t ws_size,
                              hipStream_t stream)
{
    (void)in_sizes; (void)n_in; (void)out_size; (void)ws_size;
    const int B = 8, TSRC = 256, MELMAX = 2048, D = 256, H = 2, DK = 128;
    const int DFF = 1024, NL = 4, NMELS = 80, DURF = 256;

    // ---- unpack inputs (setup_inputs dict insertion order, depth-first) ----
    int ii = 0;
    const int* src_seq = (const int*)d_in[ii++];
    const int* src_pos = (const int*)d_in[ii++];
    const int* mel_pos = (const int*)d_in[ii++];
    const int* durs    = (const int*)d_in[ii++];
    const float* src_emb = (const float*)d_in[ii++];
    const float* enc_pos = (const float*)d_in[ii++];
    const float* dec_pos = (const float*)d_in[ii++];
    FFTL enc[NL], dec[NL];
    auto grab = [&](FFTL& L) {
        L.wq   = (const float*)d_in[ii++]; L.wk   = (const float*)d_in[ii++];
        L.wv   = (const float*)d_in[ii++]; L.wo   = (const float*)d_in[ii++];
        L.bo   = (const float*)d_in[ii++];
        L.ln1g = (const float*)d_in[ii++]; L.ln1b = (const float*)d_in[ii++];
        L.c1w  = (const float*)d_in[ii++]; L.c1b  = (const float*)d_in[ii++];
        L.c2w  = (const float*)d_in[ii++]; L.c2b  = (const float*)d_in[ii++];
        L.ln2g = (const float*)d_in[ii++]; L.ln2b = (const float*)d_in[ii++];
    };
    for (int l = 0; l < NL; ++l) grab(enc[l]);
    for (int l = 0; l < NL; ++l) grab(dec[l]);
    const float* d_c1w  = (const float*)d_in[ii++];
    const float* d_c1b  = (const float*)d_in[ii++];
    const float* d_ln1g = (const float*)d_in[ii++];
    const float* d_ln1b = (const float*)d_in[ii++];
    const float* d_c2w  = (const float*)d_in[ii++];
    const float* d_c2b  = (const float*)d_in[ii++];
    const float* d_ln2g = (const float*)d_in[ii++];
    const float* d_ln2b = (const float*)d_in[ii++];
    const float* d_lw   = (const float*)d_in[ii++];
    const float* d_lb   = (const float*)d_in[ii++];
    const float* mel_w  = (const float*)d_in[ii++];
    const float* mel_b  = (const float*)d_in[ii++];

    // ---- workspace layout (f32). tmp aliases qbuf; scores alias hbuf. ----
    float* ws = (float*)d_ws;
    size_t off = 0;
    auto alloc = [&](size_t n) { float* p = ws + off; off += n; return p; };
    float* xbuf = alloc((size_t)B * MELMAX * D);       // decoder activations / y
    float* qbuf = alloc((size_t)B * MELMAX * D);
    float* kbuf = alloc((size_t)B * MELMAX * D);
    float* vbuf = alloc((size_t)B * MELMAX * D);
    float* obuf = alloc((size_t)B * MELMAX * D);
    float* hbuf = alloc((size_t)B * MELMAX * DFF);     // conv hidden; also scores
    float* encx = alloc((size_t)B * TSRC * D);
    float* dh1  = alloc((size_t)B * TSRC * DURF);
    float* dh2  = alloc((size_t)B * TSRC * DURF);
    float* tmp    = qbuf;   // qbuf dead once attention outputs are computed
    float* scores = hbuf;   // attention phase and conv phase don't overlap

    auto gemm = [&](const float* A, const float* Bm, const float* bias,
                    const float* resid, float* C,
                    int M, int N, int K, int lda, int ldb, int ldc,
                    int rsh, int rvalid,
                    long long aB, long long bB, long long cB, long long rB,
                    int Z, float alpha, int flags) {
        dim3 g((N + BN - 1) / BN, (M + BM - 1) / BM, Z);
        gemm_wmma_f16<<<g, dim3(128), 0, stream>>>(
            A, Bm, bias, resid, C, M, N, K, lda, ldb, ldc,
            rsh, rvalid, aB, bB, cB, rB, alpha, flags);
    };

    const float sc = 0.08838834764831843f;  // 1/sqrt(DK)

    // One FFT block, in place on x ([B*T, D] contiguous).
    auto fft_block = [&](float* x, int T, const FFTL& L, const int* tok) {
        int M = B * T;
        gemm(x, L.wq, nullptr, nullptr, qbuf, M, D, D, D, D, D, 0, M, 0,0,0,0, 1, 1.f, 0);
        gemm(x, L.wk, nullptr, nullptr, kbuf, M, D, D, D, D, D, 0, M, 0,0,0,0, 1, 1.f, 0);
        gemm(x, L.wv, nullptr, nullptr, vbuf, M, D, D, D, D, D, 0, M, 0,0,0,0, 1, 1.f, 0);
        for (int b = 0; b < B; ++b) {
            for (int h = 0; h < H; ++h) {
                long long base = (long long)b * T * D + (long long)h * DK;
                const float* qs = qbuf + base;
                const float* ks = kbuf + base;
                const float* vs = vbuf + base;
                // scores = (q @ k^T) / sqrt(DK)
                gemm(qs, ks, nullptr, nullptr, scores, T, T, DK, D, D, T,
                     0, T, 0,0,0,0, 1, sc, GF_BT);
                softmax_mask<<<dim3(T), dim3(256), 0, stream>>>(scores, tok + (long long)b * T, T);
                // o = probs @ v
                gemm(scores, vs, nullptr, nullptr, obuf + base, T, DK, T, T, D, D,
                     0, T, 0,0,0,0, 1, 1.f, 0);
            }
        }
        // tmp = o @ wo + bo + x ; x = LN(tmp) * non_pad
        gemm(obuf, L.wo, L.bo, x, tmp, M, D, D, D, D, D, 0, M, 0,0,0,0, 1, 1.f, 0);
        layernorm_mask<<<dim3(M), dim3(256), 0, stream>>>(tmp, x, L.ln1g, L.ln1b, tok, D);
        // conv1 (k=9, SAME): 9 accumulated per-batch GEMMs into hbuf
        for (int tap = 0; tap < 9; ++tap)
            gemm(x, L.c1w + (long long)tap * D * DFF,
                 tap == 0 ? L.c1b : nullptr, tap == 0 ? nullptr : hbuf, hbuf,
                 T, DFF, D, D, DFF, DFF, tap - 4, T,
                 (long long)T * D, 0, (long long)T * DFF, (long long)T * DFF,
                 B, 1.f, 0);
        // conv2 (k=1): relu(hbuf) @ c2w + c2b + x ; x = LN(tmp) * non_pad
        gemm(hbuf, L.c2w, L.c2b, x, tmp, M, D, DFF, DFF, D, D, 0, M, 0,0,0,0, 1, 1.f, GF_RELUA);
        layernorm_mask<<<dim3(M), dim3(256), 0, stream>>>(tmp, x, L.ln2g, L.ln2b, tok, D);
    };

    // ============================ Encoder ============================
    embed_add<<<dim3(B * TSRC), dim3(256), 0, stream>>>(src_seq, src_pos, src_emb, enc_pos, encx, D);
    for (int l = 0; l < NL; ++l) fft_block(encx, TSRC, enc[l], src_seq);

    // ====================== Duration predictor =======================
    {
        int M = B * TSRC;
        for (int tap = 0; tap < 3; ++tap)   // conv k=3, pad 1, relu after last tap
            gemm(encx, d_c1w + (long long)tap * D * DURF,
                 tap == 0 ? d_c1b : nullptr, tap == 0 ? nullptr : dh1, dh1,
                 TSRC, DURF, D, D, DURF, DURF, tap - 1, TSRC,
                 (long long)TSRC * D, 0, (long long)TSRC * DURF, (long long)TSRC * DURF,
                 B, 1.f, tap == 2 ? GF_RELUO : 0);
        layernorm_mask<<<dim3(M), dim3(256), 0, stream>>>(dh1, dh2, d_ln1g, d_ln1b, nullptr, DURF);
        for (int tap = 0; tap < 3; ++tap)
            gemm(dh2, d_c2w + (long long)tap * DURF * DURF,
                 tap == 0 ? d_c2b : nullptr, tap == 0 ? nullptr : dh1, dh1,
                 TSRC, DURF, DURF, DURF, DURF, DURF, tap - 1, TSRC,
                 (long long)TSRC * DURF, 0, (long long)TSRC * DURF, (long long)TSRC * DURF,
                 B, 1.f, tap == 2 ? GF_RELUO : 0);
        layernorm_mask<<<dim3(M), dim3(256), 0, stream>>>(dh1, dh2, d_ln2g, d_ln2b, nullptr, DURF);
        float* dur_out = (float*)d_out + (long long)B * MELMAX * NMELS;
        row_dot<<<dim3(M), dim3(256), 0, stream>>>(dh2, d_lw, d_lb, dur_out, DURF);
    }

    // ====================== Length regulator =========================
    hipMemsetAsync(xbuf, 0, (size_t)B * MELMAX * D * sizeof(float), stream);
    length_reg<<<dim3(TSRC, B), dim3(256), 0, stream>>>(durs, encx, xbuf, TSRC, MELMAX, D);
    pos_add<<<dim3(B * MELMAX), dim3(256), 0, stream>>>(mel_pos, dec_pos, xbuf, D);

    // ============================ Decoder ============================
    for (int l = 0; l < NL; ++l) fft_block(xbuf, MELMAX, dec[l], mel_pos);

    // Final mask == non_pad_mel (already applied by last LN); mel_b is still
    // added to masked rows, matching the reference (y*mask @ W + b).
    gemm(xbuf, mel_w, mel_b, nullptr, (float*)d_out,
         B * MELMAX, NMELS, D, D, NMELS, NMELS,
         0, B * MELMAX, 0,0,0,0, 1, 1.f, 0);
}
